// Conv_Attention_46763603918910
// MI455X (gfx1250) — matrix operand order repaired and checked by reading, not compiled, not run on hardware
//
#include <hip/hip_runtime.h>
#include <hip/hip_bf16.h>

// ---------------------------------------------------------------------------
// Problem constants (from reference): B=8, T=2048, Cin=512, U=512, K=3
// ---------------------------------------------------------------------------
#define BB   8
#define TT   2048
#define CIN  512
#define UU   512
#define KW   3
#define LDK  72          // padded LDS row stride (elements) for 64-wide K tiles

typedef __attribute__((ext_vector_type(16))) __bf16        bf16x16;
typedef __attribute__((ext_vector_type(8)))  float         floatx8;
typedef __attribute__((ext_vector_type(4)))  unsigned int  uintx4;

union FragU { bf16x16 v; uintx4 q[2]; };

// ---------------------------------------------------------------------------
// Fragment loads (generic pointers: work from LDS).
// A-operand (16x32): lane<16 -> K 0..7 / 16..23 ; lane>=16 -> K 8..15 / 24..31
// ---------------------------------------------------------------------------
__device__ __forceinline__ bf16x16 load_a(const __bf16* base, int ld,
                                          int row0, int k0) {
  int lane = threadIdx.x & 31;
  const __bf16* p = base + (size_t)(row0 + (lane & 15)) * ld
                         + (k0 + ((lane & 16) ? 8 : 0));
  FragU f;
  f.q[0] = *(const uintx4*)p;
  f.q[1] = *(const uintx4*)(p + 16);
  return f.v;
}

// B-operand (32x16) from TRANSPOSED storage (row n = column n of B):
// lane = N, lane<16 -> K 0..7 / 16..23, lane>=16 -> K 8..15 / 24..31
__device__ __forceinline__ bf16x16 load_bT(const __bf16* baseT, int ld,
                                           int col0, int k0) {
  int lane = threadIdx.x & 31;
  const __bf16* p = baseT + (size_t)(col0 + (lane & 15)) * ld
                          + (k0 + ((lane & 16) ? 8 : 0));
  FragU f;
  f.q[0] = *(const uintx4*)p;
  f.q[1] = *(const uintx4*)(p + 16);
  return f.v;
}

__device__ __forceinline__ floatx8 wmma_bf16(bf16x16 a, bf16x16 b, floatx8 c) {
  return __builtin_amdgcn_wmma_f32_16x16x32_bf16(
      false, a, false, b, (short)0, c, false, false);
}

// ---------------------------------------------------------------------------
// CDNA5 async global->LDS (ASYNCcnt-tracked DMA through the WGP).
// Generic LDS pointer low 32 bits == LDS byte address (aperture mapping).
// ---------------------------------------------------------------------------
__device__ __forceinline__ void async_b128(const void* src, void* ldsDst) {
  unsigned dst = (unsigned)(uintptr_t)ldsDst;
  asm volatile("global_load_async_to_lds_b128 %0, %1, off"
               :: "v"(dst), "v"(src) : "memory");
}
__device__ __forceinline__ void wait_async0() {
  asm volatile("s_wait_asynccnt 0x0" ::: "memory");
}

// ---------------------------------------------------------------------------
// Tile staging for one 64-deep K step, 256 threads.
//   A tile: 64 rows x 64 K (8 KB)  -> thread: r = t/4, 16 elems (2 x b128)
//   B tile: 128 rows x 64 K (16 KB)-> thread: r = t/2, 32 elems (4 x b128)
// ---------------------------------------------------------------------------
__device__ __forceinline__ void stageA64_async(const __bf16* gA, int ld,
                                               int row0, int k0, __bf16* lA) {
  int t = threadIdx.x, r = t >> 2, cg = (t & 3) * 16;
  const __bf16* p = gA + (size_t)(row0 + r) * ld + k0 + cg;
  __bf16* d = lA + r * LDK + cg;
  async_b128(p, d);
  async_b128(p + 8, d + 8);
}

// Zero-filling synchronous variant for causal-boundary blocks (rows < 0).
__device__ __forceinline__ void stageA64_sync_guard(const __bf16* gA, int ld,
                                                    int row0, int k0, __bf16* lA) {
  int t = threadIdx.x, r = t >> 2, cg = (t & 3) * 16;
  int grow = row0 + r;
  uintx4 v0 = {0u,0u,0u,0u}, v1 = {0u,0u,0u,0u};
  if (grow >= 0) {
    const __bf16* p = gA + (size_t)grow * ld + k0 + cg;
    v0 = *(const uintx4*)p;
    v1 = *(const uintx4*)(p + 8);
  }
  __bf16* d = lA + r * LDK + cg;
  *(uintx4*)d = v0;
  *(uintx4*)(d + 8) = v1;
}

__device__ __forceinline__ void stageB64_async(const __bf16* gBt, int ld,
                                               int col0, int k0, __bf16* lB) {
  int t = threadIdx.x, r = t >> 1, cg = (t & 1) * 32;
  const __bf16* p = gBt + (size_t)(col0 + r) * ld + k0 + cg;
  __bf16* d = lB + r * LDK + cg;
#pragma unroll
  for (int i = 0; i < 4; ++i) async_b128(p + i * 8, d + i * 8);
}

// ---------------------------------------------------------------------------
// K1: fp32 -> bf16 pack of x
// ---------------------------------------------------------------------------
__global__ void ca_pack_x(const float* __restrict__ x, __bf16* __restrict__ xb,
                          int n) {
  int i = blockIdx.x * blockDim.x + threadIdx.x;
  if (i < n) xb[i] = (__bf16)x[i];
}

// ---------------------------------------------------------------------------
// K2: W [K*Cin, U] fp32 -> Wt [U, K*Cin] bf16 (transposed)
// ---------------------------------------------------------------------------
__global__ void ca_pack_wT(const float* __restrict__ W, __bf16* __restrict__ Wt) {
  int i  = blockIdx.x * blockDim.x + threadIdx.x;   // over U * K*Cin
  int kc = i % (KW * CIN);
  int u  = i / (KW * CIN);
  Wt[i] = (__bf16)W[(size_t)kc * UU + u];
}

// ---------------------------------------------------------------------------
// K3: fused causal conv GEMM. Macro tile 64(T) x 128(U), K-step 64, double-
// buffered LDS. 8 waves, each a 32x32 register block (2x2 accumulators).
// q,k row-major bf16; v transposed [U][T] bf16.
// ---------------------------------------------------------------------------
__global__ void ca_conv_qkv(const __bf16* __restrict__ xb,
                            const __bf16* __restrict__ Wt,   // [3][U][K*Cin]
                            const float* __restrict__ bq,
                            const float* __restrict__ bk,
                            const float* __restrict__ bv,
                            __bf16* __restrict__ qb,
                            __bf16* __restrict__ kb,
                            __bf16* __restrict__ vbT) {
  __shared__ __bf16 lA[2][64 * LDK];
  __shared__ __bf16 lB[2][128 * LDK];

  int blk = blockIdx.x;
  int nt  = blk & 3;          // U tile   (512/128)
  int mt  = (blk >> 2) & 31;  // T tile   (2048/64)
  int b   = (blk >> 7) & 7;   // batch
  int which = blk >> 10;      // 0=q 1=k 2=v

  const __bf16* xB  = xb + (size_t)b * TT * CIN;
  const __bf16* WtW = Wt + (size_t)which * UU * (KW * CIN);

  int wv = threadIdx.x >> 5;
  int mo = (wv & 1) * 32, no = (wv >> 1) * 32;
  int rowBase = mt * 64, colBase = nt * 128;
  bool gblk = (mt == 0);      // only these blocks touch t<0 rows

  auto do_stageA = [&](int ks, __bf16* dst) {
    int k0 = ks * 64;
    int kk = k0 >> 9;                  // conv tap
    int c0 = k0 & (CIN - 1);
    int shift = kk - (KW - 1);         // -2,-1,0 (causal)
    if (gblk) stageA64_sync_guard(xB, CIN, rowBase + shift, c0, dst);
    else      stageA64_async(xB, CIN, rowBase + shift, c0, dst);
  };

  floatx8 z = {0.f,0.f,0.f,0.f,0.f,0.f,0.f,0.f};
  floatx8 acc00 = z, acc01 = z, acc10 = z, acc11 = z;

  const int NKS = (KW * CIN) / 64;     // 24
  do_stageA(0, lA[0]);
  stageB64_async(WtW, KW * CIN, colBase, 0, lB[0]);
  wait_async0();
  __syncthreads();

  for (int ks = 0; ks < NKS; ++ks) {
    int cur = ks & 1, nxt = cur ^ 1;
    if (ks + 1 < NKS) {
      do_stageA(ks + 1, lA[nxt]);
      stageB64_async(WtW, KW * CIN, colBase, (ks + 1) * 64, lB[nxt]);
    }
    const __bf16* A  = lA[cur];
    const __bf16* Bt = lB[cur];
#pragma unroll
    for (int kc = 0; kc < 64; kc += 32) {
      bf16x16 a0 = load_a(A, LDK, mo, kc);
      bf16x16 a1 = load_a(A, LDK, mo + 16, kc);
      bf16x16 b0 = load_bT(Bt, LDK, no, kc);
      bf16x16 b1 = load_bT(Bt, LDK, no + 16, kc);
      acc00 = wmma_bf16(a0, b0, acc00);
      acc01 = wmma_bf16(a0, b1, acc01);
      acc10 = wmma_bf16(a1, b0, acc10);
      acc11 = wmma_bf16(a1, b1, acc11);
    }
    wait_async0();
    __syncthreads();
  }

  const float* bias = (which == 0) ? bq : (which == 1) ? bk : bv;
  int lane = threadIdx.x & 31;
  int rsub = (lane & 16) ? 8 : 0;
  int csub = lane & 15;

  floatx8 accs[2][2] = {{acc00, acc01}, {acc10, acc11}};
#pragma unroll
  for (int mf = 0; mf < 2; ++mf)
#pragma unroll
    for (int nf = 0; nf < 2; ++nf) {
      int r0  = rowBase + mo + mf * 16 + rsub;
      int col = colBase + no + nf * 16 + csub;
      float bs = bias[col];
      if (which == 2) {
        __bf16* dst = vbT + (size_t)b * UU * TT;
#pragma unroll
        for (int r = 0; r < 8; ++r)
          dst[(size_t)col * TT + (r0 + r)] = (__bf16)(accs[mf][nf][r] + bs);
      } else {
        __bf16* dst = ((which == 0) ? qb : kb) + (size_t)b * TT * UU;
#pragma unroll
        for (int r = 0; r < 8; ++r)
          dst[(size_t)(r0 + r) * UU + col] = (__bf16)(accs[mf][nf][r] + bs);
      }
    }
}

// ---------------------------------------------------------------------------
// K4: S = (q @ k^T)/sqrt(U), bf16 [B][Tq][Tk]. Double-buffered async staging.
// ---------------------------------------------------------------------------
__global__ void ca_score(const __bf16* __restrict__ qb,
                         const __bf16* __restrict__ kb,
                         __bf16* __restrict__ Sb) {
  __shared__ __bf16 lA[2][64 * LDK];
  __shared__ __bf16 lB[2][128 * LDK];

  int blk = blockIdx.x;
  int nt  = blk & 15;          // Tk tile (2048/128)
  int mt  = (blk >> 4) & 31;   // Tq tile (2048/64)
  int b   = blk >> 9;

  const __bf16* qB = qb + (size_t)b * TT * UU;
  const __bf16* kB = kb + (size_t)b * TT * UU;

  int wv = threadIdx.x >> 5;
  int mo = (wv & 1) * 32, no = (wv >> 1) * 32;
  int rowBase = mt * 64, colBase = nt * 128;

  floatx8 z = {0.f,0.f,0.f,0.f,0.f,0.f,0.f,0.f};
  floatx8 acc00 = z, acc01 = z, acc10 = z, acc11 = z;

  const int NKS = UU / 64;     // 8
  stageA64_async(qB, UU, rowBase, 0, lA[0]);
  stageB64_async(kB, UU, colBase, 0, lB[0]);
  wait_async0();
  __syncthreads();

  for (int ks = 0; ks < NKS; ++ks) {
    int cur = ks & 1, nxt = cur ^ 1;
    if (ks + 1 < NKS) {
      stageA64_async(qB, UU, rowBase, (ks + 1) * 64, lA[nxt]);
      stageB64_async(kB, UU, colBase, (ks + 1) * 64, lB[nxt]);
    }
    const __bf16* A  = lA[cur];
    const __bf16* Bt = lB[cur];
#pragma unroll
    for (int kc = 0; kc < 64; kc += 32) {
      bf16x16 a0 = load_a(A, LDK, mo, kc);
      bf16x16 a1 = load_a(A, LDK, mo + 16, kc);
      bf16x16 b0 = load_bT(Bt, LDK, no, kc);
      bf16x16 b1 = load_bT(Bt, LDK, no + 16, kc);
      acc00 = wmma_bf16(a0, b0, acc00);
      acc01 = wmma_bf16(a0, b1, acc01);
      acc10 = wmma_bf16(a1, b0, acc10);
      acc11 = wmma_bf16(a1, b1, acc11);
    }
    wait_async0();
    __syncthreads();
  }

  const float scale = 0.044194173824159216f;  // 1/sqrt(512)
  int lane = threadIdx.x & 31;
  int rsub = (lane & 16) ? 8 : 0;
  int csub = lane & 15;
  __bf16* dst = Sb + (size_t)b * TT * TT;

  floatx8 accs[2][2] = {{acc00, acc01}, {acc10, acc11}};
#pragma unroll
  for (int mf = 0; mf < 2; ++mf)
#pragma unroll
    for (int nf = 0; nf < 2; ++nf) {
      int r0  = rowBase + mo + mf * 16 + rsub;
      int col = colBase + no + nf * 16 + csub;
#pragma unroll
      for (int r = 0; r < 8; ++r)
        dst[(size_t)(r0 + r) * TT + col] = (__bf16)(accs[mf][nf][r] * scale);
    }
}

// ---------------------------------------------------------------------------
// K5: per-COLUMN (softmax over query axis!) online max & sum-exp
// ---------------------------------------------------------------------------
__global__ void ca_colstats(const __bf16* __restrict__ Sb,
                            float* __restrict__ M, float* __restrict__ Z) {
  int b = blockIdx.x >> 3;
  int k = ((blockIdx.x & 7) << 8) + threadIdx.x;
  const __bf16* S = Sb + (size_t)b * TT * TT + k;
  float m = -3.4e38f, zz = 0.f;
  for (int q = 0; q < TT; ++q) {
    float s  = (float)S[(size_t)q * TT];
    float mn = fmaxf(m, s);
    zz = zz * __expf(m - mn) + __expf(s - mn);
    m = mn;
  }
  M[b * TT + k] = m;
  Z[b * TT + k] = zz;
}

// ---------------------------------------------------------------------------
// K6: P[q,k] = exp(S[q,k]-M[k]) / Z[k]  (in place, bf16)
// ---------------------------------------------------------------------------
__global__ void ca_prob(__bf16* __restrict__ Sb,
                        const float* __restrict__ M,
                        const float* __restrict__ Z) {
  size_t i = (size_t)blockIdx.x * blockDim.x + threadIdx.x;  // B*T*T
  int k = (int)(i & (TT - 1));
  int b = (int)(i >> 22);
  float s = (float)Sb[i];
  Sb[i] = (__bf16)(__expf(s - M[b * TT + k]) / Z[b * TT + k]);
}

// ---------------------------------------------------------------------------
// K7: context = P @ V (B via transposed vbT). fp32 out. Double-buffered async.
// ---------------------------------------------------------------------------
__global__ void ca_context(const __bf16* __restrict__ Pb,
                           const __bf16* __restrict__ vbT,
                           float* __restrict__ out) {
  __shared__ __bf16 lA[2][64 * LDK];
  __shared__ __bf16 lB[2][128 * LDK];

  int blk = blockIdx.x;
  int nt  = blk & 3;           // U tile (512/128)
  int mt  = (blk >> 2) & 31;   // Tq tile (2048/64)
  int b   = blk >> 7;

  const __bf16* P = Pb  + (size_t)b * TT * TT;
  const __bf16* V = vbT + (size_t)b * UU * TT;

  int wv = threadIdx.x >> 5;
  int mo = (wv & 1) * 32, no = (wv >> 1) * 32;
  int rowBase = mt * 64, colBase = nt * 128;

  floatx8 z = {0.f,0.f,0.f,0.f,0.f,0.f,0.f,0.f};
  floatx8 acc00 = z, acc01 = z, acc10 = z, acc11 = z;

  const int NKS = TT / 64;     // 32
  stageA64_async(P, TT, rowBase, 0, lA[0]);
  stageB64_async(V, TT, colBase, 0, lB[0]);
  wait_async0();
  __syncthreads();

  for (int ks = 0; ks < NKS; ++ks) {
    int cur = ks & 1, nxt = cur ^ 1;
    if (ks + 1 < NKS) {
      stageA64_async(P, TT, rowBase, (ks + 1) * 64, lA[nxt]);
      stageB64_async(V, TT, colBase, (ks + 1) * 64, lB[nxt]);
    }
    const __bf16* A  = lA[cur];
    const __bf16* Bt = lB[cur];
#pragma unroll
    for (int kc = 0; kc < 64; kc += 32) {
      bf16x16 a0 = load_a(A, LDK, mo, kc);
      bf16x16 a1 = load_a(A, LDK, mo + 16, kc);
      bf16x16 b0 = load_bT(Bt, LDK, no, kc);
      bf16x16 b1 = load_bT(Bt, LDK, no + 16, kc);
      acc00 = wmma_bf16(a0, b0, acc00);
      acc01 = wmma_bf16(a0, b1, acc01);
      acc10 = wmma_bf16(a1, b0, acc10);
      acc11 = wmma_bf16(a1, b1, acc11);
    }
    wait_async0();
    __syncthreads();
  }

  int lane = threadIdx.x & 31;
  int rsub = (lane & 16) ? 8 : 0;
  int csub = lane & 15;
  float* dst = out + (size_t)b * TT * UU;

  floatx8 accs[2][2] = {{acc00, acc01}, {acc10, acc11}};
#pragma unroll
  for (int mf = 0; mf < 2; ++mf)
#pragma unroll
    for (int nf = 0; nf < 2; ++nf) {
      int r0  = rowBase + mo + mf * 16 + rsub;
      int col = colBase + no + nf * 16 + csub;
#pragma unroll
      for (int r = 0; r < 8; ++r)
        dst[(size_t)(r0 + r) * UU + col] = accs[mf][nf][r];
    }
}

// ---------------------------------------------------------------------------
// Host-side launcher
// ---------------------------------------------------------------------------
extern "C" void kernel_launch(void* const* d_in, const int* in_sizes, int n_in,
                              void* d_out, int out_size, void* d_ws, size_t ws_size,
                              hipStream_t stream) {
  (void)in_sizes; (void)n_in; (void)out_size; (void)ws_size;

  const float* x  = (const float*)d_in[0];
  const float* Wq = (const float*)d_in[1];
  const float* bq = (const float*)d_in[2];
  const float* Wk = (const float*)d_in[3];
  const float* bk = (const float*)d_in[4];
  const float* Wv = (const float*)d_in[5];
  const float* bv = (const float*)d_in[6];
  float* out = (float*)d_out;

  char* w = (char*)d_ws;
  __bf16* xb  = (__bf16*)w; w += (size_t)BB * TT * CIN * 2;        // 16 MiB
  __bf16* Wt  = (__bf16*)w; w += (size_t)3 * UU * KW * CIN * 2;    // 4.5 MiB
  __bf16* qb  = (__bf16*)w; w += (size_t)BB * TT * UU * 2;         // 16 MiB
  __bf16* kb  = (__bf16*)w; w += (size_t)BB * TT * UU * 2;         // 16 MiB
  __bf16* vbT = (__bf16*)w; w += (size_t)BB * UU * TT * 2;         // 16 MiB
  __bf16* Sb  = (__bf16*)w; w += (size_t)BB * TT * TT * 2;         // 64 MiB
  float*  M   = (float*)w;  w += (size_t)BB * TT * 4;
  float*  Z   = (float*)w;

  const int nx = BB * TT * CIN;                    // 8,388,608
  ca_pack_x<<<nx / 256, 256, 0, stream>>>(x, xb, nx);

  const int nw = UU * KW * CIN;                    // 786,432 per weight
  ca_pack_wT<<<nw / 256, 256, 0, stream>>>(Wq, Wt);
  ca_pack_wT<<<nw / 256, 256, 0, stream>>>(Wk, Wt + (size_t)nw);
  ca_pack_wT<<<nw / 256, 256, 0, stream>>>(Wv, Wt + (size_t)2 * nw);

  // conv: 3 outs * 8 batches * 32 M-tiles(64) * 4 N-tiles(128)
  ca_conv_qkv<<<3 * BB * 32 * 4, 256, 0, stream>>>(xb, Wt, bq, bk, bv,
                                                   qb, kb, vbT);
  // score: 8 * 32 * 16
  ca_score<<<BB * 32 * 16, 256, 0, stream>>>(qb, kb, Sb);

  ca_colstats<<<BB * (TT / 256), 256, 0, stream>>>(Sb, M, Z);

  ca_prob<<<(int)(((size_t)BB * TT * TT) / 256), 256, 0, stream>>>(Sb, M, Z);

  // context: 8 * 32 * 4
  ca_context<<<BB * 32 * 4, 256, 0, stream>>>(Sb, vbT, out);
}
